// GNNProcessor_37984690765827
// MI455X (gfx1250) — matrix operand order repaired and checked by reading, not compiled, not run on hardware
//
#include <hip/hip_runtime.h>
#include <hip/hip_bf16.h>

// ---------------------------------------------------------------------------
// GNN processor for MI455X (gfx1250, wave32).
// Compute-bound on matrix throughput (~165 GFLOP vs ~0.5 GB of traffic),
// so all GEMM work uses v_wmma_f32_16x16x32_f16 (f16 in, f32 accumulate).
// Each wave computes a 16-row x 128-col tile (8 accumulator tiles of 16x16).
// Weights are pre-packed into WMMA B-fragment order (f16) in workspace.
// ---------------------------------------------------------------------------

#define HD    128   // hidden dim
#define NT    8     // 128 / 16 column tiles
#define WAVES 8     // waves per 256-thread block
#define LN_EPS 1e-5f

typedef __attribute__((ext_vector_type(16))) _Float16 v16h;
typedef __attribute__((ext_vector_type(8)))  _Float16 v8h;
typedef __attribute__((ext_vector_type(8)))  float    v8f;

static __device__ inline v8f vzero() {
  v8f z;
#pragma unroll
  for (int i = 0; i < 8; ++i) z[i] = 0.0f;
  return z;
}

// One A-fragment (16x32 f16) x full row of B tiles -> 8 WMMAs.
// B fragments are packed so each lane reads 16 contiguous halves (32B).
static __device__ inline void wmma_row(v16h a, const _Float16* __restrict__ Wp,
                                       int fragBase, int lane, v8f acc[NT]) {
#pragma unroll
  for (int t = 0; t < NT; ++t) {
    v16h b = *(const v16h*)(Wp + ((size_t)(fragBase + t) * 32 + lane) * 16);
    acc[t] = __builtin_amdgcn_wmma_f32_16x16x32_f16(
        false, a, false, b, (short)0, acc[t], false, false);
  }
}

// A-fragment from a row-major f32 row in global memory.
// Layout (16-bit A 16x32): lane<16 holds K {k0..k0+7, k0+16..k0+23},
// lane>=16 holds K {k0+8..k0+15, k0+24..k0+31}; caller folds hi*8 into k0.
static __device__ inline v16h afrag_global(const float* __restrict__ base, int k0) {
  v16h a;
#pragma unroll
  for (int i = 0; i < 8; ++i) a[i] = (_Float16)base[k0 + i];
#pragma unroll
  for (int i = 0; i < 8; ++i) a[8 + i] = (_Float16)base[k0 + 16 + i];
  return a;
}

// 16x128 GEMM stage reading the f16 activation tile from LDS (K = 128).
static __device__ inline void gemm_lds(const _Float16* Arow,
                                       const _Float16* __restrict__ Wp,
                                       int lane, v8f acc[NT]) {
  const int hi = lane >> 4;
#pragma unroll
  for (int c = 0; c < 4; ++c) {
    if (c < 3)  // warm next weight chunk into cache (global_prefetch)
      __builtin_prefetch(Wp + ((size_t)((c + 1) * NT) * 32 + lane) * 16, 0, 1);
    const int k0 = c * 32 + hi * 8;
    v8h s0 = *(const v8h*)(Arow + k0);
    v8h s1 = *(const v8h*)(Arow + k0 + 16);
    v16h a;
#pragma unroll
    for (int i = 0; i < 8; ++i) { a[i] = s0[i]; a[8 + i] = s1[i]; }
    wmma_row(a, Wp, c * NT, lane, acc);
  }
}

// bias + SiLU, store f16 activation tile to this wave's LDS region.
// C layout: lane<16 -> col=lane rows 0..7 (vgpr r); lane>=16 -> rows 8..15.
static __device__ inline void silu_store(const v8f acc[NT],
                                         const float* __restrict__ bias,
                                         _Float16* Abuf, int lane) {
  const int n0 = lane & 15, hi = lane >> 4;
#pragma unroll
  for (int t = 0; t < NT; ++t) {
    const int col = t * 16 + n0;
    const float bb = bias[col];
#pragma unroll
    for (int r = 0; r < 8; ++r) {
      float v = acc[t][r] + bb;
      v = v / (1.0f + __expf(-v));
      Abuf[(size_t)(r + 8 * hi) * HD + col] = (_Float16)v;
    }
  }
}

// bias + LayerNorm over the 128-wide rows, fully in registers.
// A row's 128 values live in one 16-lane half (8 values per lane across the
// 8 column tiles); reduce with xor-butterflies that stay inside the half.
static __device__ inline void ln_epilogue(v8f acc[NT],
                                          const float* __restrict__ bias,
                                          const float* __restrict__ g,
                                          const float* __restrict__ be,
                                          int lane) {
  const int n0 = lane & 15;
  float cb[NT], cg[NT], cbe[NT];
#pragma unroll
  for (int t = 0; t < NT; ++t) {
    const int col = t * 16 + n0;
    cb[t] = bias[col]; cg[t] = g[col]; cbe[t] = be[col];
  }
#pragma unroll
  for (int t = 0; t < NT; ++t)
#pragma unroll
    for (int r = 0; r < 8; ++r) acc[t][r] += cb[t];
#pragma unroll
  for (int r = 0; r < 8; ++r) {
    float s = 0.0f;
#pragma unroll
    for (int t = 0; t < NT; ++t) s += acc[t][r];
    s += __shfl_xor(s, 1); s += __shfl_xor(s, 2);
    s += __shfl_xor(s, 4); s += __shfl_xor(s, 8);
    const float mu = s * (1.0f / HD);
    float q = 0.0f;
#pragma unroll
    for (int t = 0; t < NT; ++t) { const float d = acc[t][r] - mu; q += d * d; }
    q += __shfl_xor(q, 1); q += __shfl_xor(q, 2);
    q += __shfl_xor(q, 4); q += __shfl_xor(q, 8);
    const float rstd = rsqrtf(q * (1.0f / HD) + LN_EPS);
#pragma unroll
    for (int t = 0; t < NT; ++t)
      acc[t][r] = (acc[t][r] - mu) * rstd * cg[t] + cbe[t];
  }
}

// ---------------------------------------------------------------------------
// Pack an f32 weight matrix [K x N] (row-major, k-major as in the reference)
// into WMMA B-fragment order (f16): frag (c,t) = K-chunk c, N-tile t; within a
// fragment, lane holds 16 contiguous halves: k = c*32 + (lane>>4)*8 + (i&7) + (i>>3)*16,
// n = t*16 + (lane&15). Zero-pads K up to a multiple of 32 (emb_W1: K=16).
// ---------------------------------------------------------------------------
__global__ void pack_w_kernel(const float* __restrict__ src,
                              _Float16* __restrict__ dst,
                              int K, int Kc, int N) {
  const int tid = blockIdx.x * blockDim.x + threadIdx.x;
  const int NTl = N / 16;
  const int total = Kc * NTl * 512;
  if (tid >= total) return;
  const int i    = tid & 15;
  const int lane = (tid >> 4) & 31;
  const int frag = tid >> 9;
  const int c = frag / NTl, t = frag % NTl;
  const int n = t * 16 + (lane & 15);
  const int k = c * 32 + (lane >> 4) * 8 + (i & 7) + (i >> 3) * 16;
  dst[tid] = (k < K) ? (_Float16)src[(size_t)k * N + n] : (_Float16)0.0f;
}

__global__ void zero_kernel(float* __restrict__ p, int n) {
  const int i = blockIdx.x * blockDim.x + threadIdx.x;
  if (i < n) p[i] = 0.0f;
}

// ---------------------------------------------------------------------------
// Edge-embedding MLP: e = LN(MLP(edge_attr)).  K1 = 16 (padded to 32).
// ---------------------------------------------------------------------------
__global__ __launch_bounds__(256) void emb_kernel(
    const float* __restrict__ eattr, float* __restrict__ e_out,
    const _Float16* __restrict__ W1, const float* __restrict__ b1,
    const _Float16* __restrict__ W2, const float* __restrict__ b2,
    const _Float16* __restrict__ W3, const float* __restrict__ b3,
    const float* __restrict__ g, const float* __restrict__ be, int nE) {
  __shared__ _Float16 hbuf[WAVES][16 * HD];
  const int lane = threadIdx.x & 31;
  const int w    = threadIdx.x >> 5;
  const long tile = (long)blockIdx.x * WAVES + w;
  const long rowbase = tile * 16;
  if (rowbase >= nE) return;
  const int m = lane & 15, hi = lane >> 4;
  const long arow = rowbase + m;
  const float* row = eattr + (size_t)arow * 16;

  v8f acc[NT];
#pragma unroll
  for (int t = 0; t < NT; ++t) acc[t] = vzero();

  // layer 1: K=16, one chunk, second K-half is zero padding
  v16h a;
#pragma unroll
  for (int i = 0; i < 8; ++i) a[i] = (_Float16)row[hi * 8 + i];
#pragma unroll
  for (int i = 0; i < 8; ++i) a[8 + i] = (_Float16)0.0f;
  wmma_row(a, W1, 0, lane, acc);

  _Float16* Abuf = hbuf[w];
  silu_store(acc, b1, Abuf, lane);

#pragma unroll
  for (int t = 0; t < NT; ++t) acc[t] = vzero();
  gemm_lds(Abuf + (size_t)m * HD, W2, lane, acc);
  silu_store(acc, b2, Abuf, lane);

#pragma unroll
  for (int t = 0; t < NT; ++t) acc[t] = vzero();
  gemm_lds(Abuf + (size_t)m * HD, W3, lane, acc);
  ln_epilogue(acc, b3, g, be, lane);

  const int n0 = lane & 15;
#pragma unroll
  for (int t = 0; t < NT; ++t) {
    const int col = t * 16 + n0;
#pragma unroll
    for (int r = 0; r < 8; ++r) {
      const long orow = rowbase + r + 8 * hi;
      e_out[(size_t)orow * HD + col] = acc[t][r];
    }
  }
}

// ---------------------------------------------------------------------------
// Edge block: e_new = LN(MLP([x[dst], x[src], e])) + e ; agg[dst] += e_new.
// K1 = 384 (12 chunks of 32).
// ---------------------------------------------------------------------------
__global__ __launch_bounds__(256) void edge_block_kernel(
    const float* __restrict__ x,
    const long long* __restrict__ srcI, const long long* __restrict__ dstI,
    const float* __restrict__ e_in, float* __restrict__ e_out,
    float* __restrict__ agg,
    const _Float16* __restrict__ W1, const float* __restrict__ b1,
    const _Float16* __restrict__ W2, const float* __restrict__ b2,
    const _Float16* __restrict__ W3, const float* __restrict__ b3,
    const float* __restrict__ g, const float* __restrict__ be, int nE) {
  __shared__ _Float16 hbuf[WAVES][16 * HD];
  const int lane = threadIdx.x & 31;
  const int w    = threadIdx.x >> 5;
  const long tile = (long)blockIdx.x * WAVES + w;
  const long rowbase = tile * 16;
  if (rowbase >= nE) return;
  const int m = lane & 15, hi = lane >> 4;
  const long arow = rowbase + m;
  const long long sN = srcI[arow];
  const long long dN = dstI[arow];
  const float* pxi = x + (size_t)dN * HD;        // x_i = x[dst]
  const float* pxj = x + (size_t)sN * HD;        // x_j = x[src]
  const float* pe  = e_in + (size_t)arow * HD;   // e

  v8f acc[NT];
#pragma unroll
  for (int t = 0; t < NT; ++t) acc[t] = vzero();

  // layer 1 over concat [x_i | x_j | e], K = 384
#pragma unroll
  for (int c = 0; c < 12; ++c) {
    const float* base = (c < 4) ? pxi : (c < 8) ? pxj : pe;
    const int k0 = (c & 3) * 32 + hi * 8;
    wmma_row(afrag_global(base, k0), W1, c * NT, lane, acc);
  }
  _Float16* Abuf = hbuf[w];
  silu_store(acc, b1, Abuf, lane);

#pragma unroll
  for (int t = 0; t < NT; ++t) acc[t] = vzero();
  gemm_lds(Abuf + (size_t)m * HD, W2, lane, acc);
  silu_store(acc, b2, Abuf, lane);

#pragma unroll
  for (int t = 0; t < NT; ++t) acc[t] = vzero();
  gemm_lds(Abuf + (size_t)m * HD, W3, lane, acc);
  ln_epilogue(acc, b3, g, be, lane);

  // residual, store e_new, and scatter-add into agg[dst]
  long long dn8[8];
#pragma unroll
  for (int r = 0; r < 8; ++r) dn8[r] = dstI[rowbase + r + 8 * hi];
  const int n0 = lane & 15;
#pragma unroll
  for (int t = 0; t < NT; ++t) {
    const int col = t * 16 + n0;
#pragma unroll
    for (int r = 0; r < 8; ++r) {
      const long orow = rowbase + r + 8 * hi;
      const float v = acc[t][r] + e_in[(size_t)orow * HD + col];
      e_out[(size_t)orow * HD + col] = v;
      atomicAdd(&agg[(size_t)dn8[r] * HD + col], v);
    }
  }
}

// ---------------------------------------------------------------------------
// Node block: x_new = LN(MLP([x | agg])) + x.  K1 = 256 (8 chunks).
// ---------------------------------------------------------------------------
__global__ __launch_bounds__(256) void node_block_kernel(
    const float* __restrict__ x_in, const float* __restrict__ agg,
    float* __restrict__ x_out,
    const _Float16* __restrict__ W1, const float* __restrict__ b1,
    const _Float16* __restrict__ W2, const float* __restrict__ b2,
    const _Float16* __restrict__ W3, const float* __restrict__ b3,
    const float* __restrict__ g, const float* __restrict__ be, int nN) {
  __shared__ _Float16 hbuf[WAVES][16 * HD];
  const int lane = threadIdx.x & 31;
  const int w    = threadIdx.x >> 5;
  const long tile = (long)blockIdx.x * WAVES + w;
  const long rowbase = tile * 16;
  if (rowbase >= nN) return;
  const int m = lane & 15, hi = lane >> 4;
  long arow = rowbase + m;
  if (arow >= nN) arow = nN - 1;  // clamp gathers; stores are masked below
  const float* px = x_in + (size_t)arow * HD;
  const float* pa = agg  + (size_t)arow * HD;

  v8f acc[NT];
#pragma unroll
  for (int t = 0; t < NT; ++t) acc[t] = vzero();

  // layer 1 over concat [x | agg], K = 256
#pragma unroll
  for (int c = 0; c < 8; ++c) {
    const float* base = (c < 4) ? px : pa;
    const int k0 = (c & 3) * 32 + hi * 8;
    wmma_row(afrag_global(base, k0), W1, c * NT, lane, acc);
  }
  _Float16* Abuf = hbuf[w];
  silu_store(acc, b1, Abuf, lane);

#pragma unroll
  for (int t = 0; t < NT; ++t) acc[t] = vzero();
  gemm_lds(Abuf + (size_t)m * HD, W2, lane, acc);
  silu_store(acc, b2, Abuf, lane);

#pragma unroll
  for (int t = 0; t < NT; ++t) acc[t] = vzero();
  gemm_lds(Abuf + (size_t)m * HD, W3, lane, acc);
  ln_epilogue(acc, b3, g, be, lane);

  const int n0 = lane & 15;
#pragma unroll
  for (int t = 0; t < NT; ++t) {
    const int col = t * 16 + n0;
#pragma unroll
    for (int r = 0; r < 8; ++r) {
      const long orow = rowbase + r + 8 * hi;
      if (orow < nN)
        x_out[(size_t)orow * HD + col] =
            acc[t][r] + x_in[(size_t)orow * HD + col];
    }
  }
}

// ---------------------------------------------------------------------------
// Host launcher.  Workspace layout (needs ~231 MB):
//   [ e_tmp: E*H f32 ][ agg: N*H f32 ][ x_mid: N*H f32 ][ packed f16 weights ]
// Buffer ping-pong puts final e and x directly in d_out.
// ---------------------------------------------------------------------------
extern "C" void kernel_launch(void* const* d_in, const int* in_sizes, int n_in,
                              void* d_out, int out_size, void* d_ws, size_t ws_size,
                              hipStream_t stream) {
  const float*     x     = (const float*)d_in[0];
  const long long* ei    = (const long long*)d_in[1];
  const float*     eattr = (const float*)d_in[2];
  const float* emb_W1 = (const float*)d_in[3];
  const float* emb_b1 = (const float*)d_in[4];
  const float* emb_W2 = (const float*)d_in[5];
  const float* emb_b2 = (const float*)d_in[6];
  const float* emb_W3 = (const float*)d_in[7];
  const float* emb_b3 = (const float*)d_in[8];
  const float* emb_g  = (const float*)d_in[9];
  const float* emb_be = (const float*)d_in[10];
  const float* edge_W1 = (const float*)d_in[11];
  const float* edge_b1 = (const float*)d_in[12];
  const float* edge_W2 = (const float*)d_in[13];
  const float* edge_b2 = (const float*)d_in[14];
  const float* edge_W3 = (const float*)d_in[15];
  const float* edge_b3 = (const float*)d_in[16];
  const float* edge_g  = (const float*)d_in[17];
  const float* edge_be = (const float*)d_in[18];
  const float* node_W1 = (const float*)d_in[19];
  const float* node_b1 = (const float*)d_in[20];
  const float* node_W2 = (const float*)d_in[21];
  const float* node_b2 = (const float*)d_in[22];
  const float* node_W3 = (const float*)d_in[23];
  const float* node_b3 = (const float*)d_in[24];
  const float* node_g  = (const float*)d_in[25];
  const float* node_be = (const float*)d_in[26];

  const int N = in_sizes[0] / HD;   // 25000
  const int E = in_sizes[1] / 2;    // 400000
  const long long* srcI = ei;
  const long long* dstI = ei + (size_t)E;

  float* out_x = (float*)d_out;
  float* out_e = out_x + (size_t)N * HD;

  float* eB   = (float*)d_ws;                       // E*H  temp e buffer
  float* aggb = eB   + (size_t)E * HD;              // N*H  segment-sum
  float* xmid = aggb + (size_t)N * HD;              // N*H  mid x
  _Float16* wp = (_Float16*)(xmid + (size_t)N * HD);

  // packed weight regions (sizes in halves: Kc * 8 tiles * 512)
  _Float16* embW1p  = wp;               wp += (size_t)1  * NT * 512;
  _Float16* embW2p  = wp;               wp += (size_t)4  * NT * 512;
  _Float16* embW3p  = wp;               wp += (size_t)4  * NT * 512;
  _Float16* edgeW1p[2]; _Float16* edgeW2p[2]; _Float16* edgeW3p[2];
  _Float16* nodeW1p[2]; _Float16* nodeW2p[2]; _Float16* nodeW3p[2];
  for (int l = 0; l < 2; ++l) { edgeW1p[l] = wp; wp += (size_t)12 * NT * 512; }
  for (int l = 0; l < 2; ++l) { edgeW2p[l] = wp; wp += (size_t)4  * NT * 512; }
  for (int l = 0; l < 2; ++l) { edgeW3p[l] = wp; wp += (size_t)4  * NT * 512; }
  for (int l = 0; l < 2; ++l) { nodeW1p[l] = wp; wp += (size_t)8  * NT * 512; }
  for (int l = 0; l < 2; ++l) { nodeW2p[l] = wp; wp += (size_t)4  * NT * 512; }
  for (int l = 0; l < 2; ++l) { nodeW3p[l] = wp; wp += (size_t)4  * NT * 512; }

  auto packW = [&](const float* s, _Float16* dst, int K, int Kc) {
    const int total = Kc * NT * 512;
    pack_w_kernel<<<(total + 255) / 256, 256, 0, stream>>>(s, dst, K, Kc, HD);
  };

  packW(emb_W1, embW1p, 16, 1);
  packW(emb_W2, embW2p, HD, 4);
  packW(emb_W3, embW3p, HD, 4);
  for (int l = 0; l < 2; ++l) {
    packW(edge_W1 + (size_t)l * 3 * HD * HD, edgeW1p[l], 3 * HD, 12);
    packW(edge_W2 + (size_t)l * HD * HD,     edgeW2p[l], HD, 4);
    packW(edge_W3 + (size_t)l * HD * HD,     edgeW3p[l], HD, 4);
    packW(node_W1 + (size_t)l * 2 * HD * HD, nodeW1p[l], 2 * HD, 8);
    packW(node_W2 + (size_t)l * HD * HD,     nodeW2p[l], HD, 4);
    packW(node_W3 + (size_t)l * HD * HD,     nodeW3p[l], HD, 4);
  }

  const int gridE = (E / 16 + WAVES - 1) / WAVES;            // 3125
  const int gridN = ((N + 15) / 16 + WAVES - 1) / WAVES;     // 196
  const int gridZ = (N * HD + 255) / 256;

  // e ping-pong: emb -> out_e, block0 -> eB, block1 -> out_e (final).
  emb_kernel<<<gridE, 256, 0, stream>>>(eattr, out_e,
      embW1p, emb_b1, embW2p, emb_b2, embW3p, emb_b3, emb_g, emb_be, E);

  // ---- block 0 ----
  zero_kernel<<<gridZ, 256, 0, stream>>>(aggb, N * HD);
  edge_block_kernel<<<gridE, 256, 0, stream>>>(x, srcI, dstI, out_e, eB, aggb,
      edgeW1p[0], edge_b1, edgeW2p[0], edge_b2, edgeW3p[0], edge_b3,
      edge_g, edge_be, E);
  node_block_kernel<<<gridN, 256, 0, stream>>>(x, aggb, xmid,
      nodeW1p[0], node_b1, nodeW2p[0], node_b2, nodeW3p[0], node_b3,
      node_g, node_be, N);

  // ---- block 1 ----
  zero_kernel<<<gridZ, 256, 0, stream>>>(aggb, N * HD);
  edge_block_kernel<<<gridE, 256, 0, stream>>>(xmid, srcI, dstI, eB, out_e, aggb,
      edgeW1p[1], edge_b1 + HD, edgeW2p[1], edge_b2 + HD,
      edgeW3p[1], edge_b3 + HD, edge_g + HD, edge_be + HD, E);
  node_block_kernel<<<gridN, 256, 0, stream>>>(xmid, aggb, out_x,
      nodeW1p[1], node_b1 + HD, nodeW2p[1], node_b2 + HD,
      nodeW3p[1], node_b3 + HD, node_g + HD, node_be + HD, N);
}
